// MultiheadAttentionWithRope_48223892799980
// MI455X (gfx1250) — matrix operand order repaired and checked by reading, not compiled, not run on hardware
//
#include <hip/hip_runtime.h>

typedef __bf16 bf16;
typedef __attribute__((ext_vector_type(16))) __bf16 v16bf;
typedef __attribute__((ext_vector_type(8)))  __bf16 v8bf;
typedef __attribute__((ext_vector_type(8)))  float  v8f;

#define WMMA_BF16(a, b, c) \
  __builtin_amdgcn_wmma_f32_16x16x32_bf16(false, (a), false, (b), (short)0, (c), false, false)

#define BATCH 8
#define SEQ   1024
#define DIMC  1408
#define HEADS 16
#define HD    88
#define HDP   96   // head dim padded to 3 * 32 for K, 6 * 16 for N

// ---------- WMMA fragment loaders (wave32, 16x16x32 bf16) ----------
// A fragment: 16xK tile, row-major source with leading dim lda (elements).
// lane L: row = L%16 ; elements e=0..7 -> K = (L/16)*8 + e ; e=8..15 -> K = 16+(L/16)*8+(e-8)
static __device__ __forceinline__ v16bf load_a_frag(const bf16* base, int lda) {
  const int lane = threadIdx.x & 31;
  const bf16* p = base + (size_t)(lane & 15) * lda + ((lane >> 4) * 8);
  v8bf lo = *(const v8bf*)(p);
  v8bf hi = *(const v8bf*)(p + 16);
  v16bf r;
#pragma unroll
  for (int i = 0; i < 8; ++i) { r[i] = lo[i]; r[i + 8] = hi[i]; }
  return r;
}

// B fragment: KxN tile, source stored [N][K] row-major (leading dim ldb).
// lane L: col n = L%16 ; elements e=0..7 -> K = (L/16)*8 + e ; e=8..15 -> K = 16+(L/16)*8+(e-8)
static __device__ __forceinline__ v16bf load_b_frag(const bf16* base, int ldb) {
  const int lane = threadIdx.x & 31;
  const bf16* p = base + (size_t)(lane & 15) * ldb + ((lane >> 4) * 8);
  v8bf lo = *(const v8bf*)(p);
  v8bf hi = *(const v8bf*)(p + 16);
  v16bf r;
#pragma unroll
  for (int i = 0; i < 8; ++i) { r[i] = lo[i]; r[i + 8] = hi[i]; }
  return r;
}

// ---------- fp32 -> bf16 convert ----------
__global__ void cvt_f32_bf16(const float* __restrict__ src, bf16* __restrict__ dst, int n) {
  int i = blockIdx.x * blockDim.x + threadIdx.x;
  if (i < n) dst[i] = (bf16)src[i];
}

// ---------- GEMM: C[M,N] = A[M,K] * W[N,K]^T (+bias), bf16 in, f32 out ----------
// Block: 256 threads = 8 waves as 4(M) x 2(N); wave tile 32x64 -> block tile 128x128.
// Per 32-wide k-step: 8 WMMAs per 6 fragment loads (~22 FLOP/B of fragment traffic).
__global__ __launch_bounds__(256) void gemm_bf16_wmma(
    const bf16* __restrict__ A, const bf16* __restrict__ W,
    float* __restrict__ C, const float* __restrict__ bias,
    int K, int lda, int ldb, int ldc) {
  const int w = threadIdx.x >> 5;
  const int lane = threadIdx.x & 31;
  const int mbase = blockIdx.y * 128 + (w >> 1) * 32;   // 4 waves along M
  const int nbase = blockIdx.x * 128 + (w & 1) * 64;    // 2 waves along N
  v8f acc[2][4] = {};
  const bf16* Ab[2];
  const bf16* Wb[4];
#pragma unroll
  for (int i = 0; i < 2; ++i) Ab[i] = A + (size_t)(mbase + i * 16) * lda;
#pragma unroll
  for (int j = 0; j < 4; ++j) Wb[j] = W + (size_t)(nbase + j * 16) * ldb;

  for (int k0 = 0; k0 < K; k0 += 32) {
    __builtin_prefetch(Ab[0] + k0 + 128, 0, 1);
    __builtin_prefetch(Wb[0] + k0 + 128, 0, 1);
    v16bf a[2], b[4];
#pragma unroll
    for (int i = 0; i < 2; ++i) a[i] = load_a_frag(Ab[i] + k0, lda);
#pragma unroll
    for (int j = 0; j < 4; ++j) b[j] = load_b_frag(Wb[j] + k0, ldb);
#pragma unroll
    for (int i = 0; i < 2; ++i)
#pragma unroll
      for (int j = 0; j < 4; ++j)
        acc[i][j] = WMMA_BF16(a[i], b[j], acc[i][j]);
  }
  // C/D layout: VGPR r -> row = r + 8*(lane/16), col = lane%16
  const int rbase = (lane >> 4) * 8;
  const int ncol = lane & 15;
#pragma unroll
  for (int i = 0; i < 2; ++i)
#pragma unroll
    for (int j = 0; j < 4; ++j) {
      const int col = nbase + j * 16 + ncol;
      const float bv = bias ? bias[col] : 0.0f;
#pragma unroll
      for (int r = 0; r < 8; ++r) {
        const int row = mbase + i * 16 + rbase + r;
        C[(size_t)row * ldc + col] = acc[i][j][r] + bv;
      }
    }
}

// ---------- fused RoPE + RMSNorm (+q scale) + V-transpose pack ----------
// One wave per (b,h,n). qs/ks: [B,H,N,96] bf16 (zero padded). vt: [B,H,96,N] bf16.
__global__ __launch_bounds__(256) void rope_rms_kernel(
    const float* __restrict__ qkv, const float* __restrict__ qw, const float* __restrict__ kw,
    bf16* __restrict__ qs, bf16* __restrict__ ks, bf16* __restrict__ vt) {
  const int lane = threadIdx.x & 31;
  const int wid = blockIdx.x * 8 + (threadIdx.x >> 5);   // (b*16 + h)*1024 + n
  const int n = wid & (SEQ - 1);
  const int h = (wid >> 10) & (HEADS - 1);
  const int b = wid >> 14;
  const float* qrow = qkv + (size_t)(b * SEQ + n) * (3 * DIMC) + h * HD;
  const float* krow = qrow + DIMC;
  const float* vrow = qrow + 2 * DIMC;
  const size_t obase = ((size_t)(b * HEADS + h) * SEQ + n) * HDP;

  float q0[2] = {0.f, 0.f}, q1[2] = {0.f, 0.f}, k0[2] = {0.f, 0.f}, k1[2] = {0.f, 0.f};
  float sq = 0.f, sk = 0.f;
#pragma unroll
  for (int t = 0; t < 2; ++t) {
    const int p = lane + t * 32;           // rope pair index, 44 pairs
    if (p < HD / 2) {
      const float ang = (float)n * __powf(10000.0f, -(2.0f * p) * (1.0f / (float)HD));
      float ss, cc;
      __sincosf(ang, &ss, &cc);
      float x0 = qrow[2 * p], x1 = qrow[2 * p + 1];
      q0[t] = x0 * cc - x1 * ss; q1[t] = x1 * cc + x0 * ss;
      sq += q0[t] * q0[t] + q1[t] * q1[t];
      x0 = krow[2 * p]; x1 = krow[2 * p + 1];
      k0[t] = x0 * cc - x1 * ss; k1[t] = x1 * cc + x0 * ss;
      sk += k0[t] * k0[t] + k1[t] * k1[t];
    }
  }
#pragma unroll
  for (int m = 16; m >= 1; m >>= 1) {
    sq += __shfl_xor(sq, m, 32);
    sk += __shfl_xor(sk, m, 32);
  }
  const float SCALE = 0.10660035817780521f;  // 88^-0.5, folded into q
  const float iq = rsqrtf(sq * (1.0f / (float)HD) + 1e-6f) * SCALE;
  const float ik = rsqrtf(sk * (1.0f / (float)HD) + 1e-6f);
#pragma unroll
  for (int t = 0; t < 2; ++t) {
    const int p = lane + t * 32;
    if (p < HD / 2) {
      qs[obase + 2 * p]     = (bf16)(q0[t] * iq * qw[2 * p]);
      qs[obase + 2 * p + 1] = (bf16)(q1[t] * iq * qw[2 * p + 1]);
      ks[obase + 2 * p]     = (bf16)(k0[t] * ik * kw[2 * p]);
      ks[obase + 2 * p + 1] = (bf16)(k1[t] * ik * kw[2 * p + 1]);
    }
  }
  if (lane < HDP - HD) { qs[obase + HD + lane] = (bf16)0.f; ks[obase + HD + lane] = (bf16)0.f; }
  const size_t vbase = (size_t)(b * HEADS + h) * HDP * SEQ;
  for (int d = lane; d < HD; d += 32) vt[vbase + (size_t)d * SEQ + n] = (bf16)vrow[d];
  if (lane < HDP - HD) vt[vbase + (size_t)(HD + lane) * SEQ + n] = (bf16)0.f;
}

// ---------- flash attention: one wave per 16-row query tile ----------
__global__ __launch_bounds__(256) void attn_wmma(
    const bf16* __restrict__ qs, const bf16* __restrict__ ks,
    const bf16* __restrict__ vt, bf16* __restrict__ outb) {
  __shared__ __align__(16) bf16 pstage[8][16 * 32];   // per-wave P tile (C-layout -> A-layout)
  const int w = threadIdx.x >> 5;
  const int lane = threadIdx.x & 31;
  const int id = blockIdx.x * 8 + w;
  const int qt = id & 63;
  const int h = (id >> 6) & (HEADS - 1);
  const int b = id >> 10;
  const size_t rowbase = (size_t)(b * HEADS + h) * SEQ;
  const bf16* qbase = qs + (rowbase + (size_t)qt * 16) * HDP;

  v16bf qf[3];
#pragma unroll
  for (int c = 0; c < 3; ++c) qf[c] = load_a_frag(qbase + c * 32, HDP);

  v8f o[6] = {};
  float rmax[8], rsum[8];
#pragma unroll
  for (int r = 0; r < 8; ++r) { rmax[r] = -1e30f; rsum[r] = 0.f; }

  const bf16* kbb = ks + rowbase * HDP;
  const bf16* vbb = vt + (size_t)(b * HEADS + h) * HDP * SEQ;

  for (int kt = 0; kt < SEQ; kt += 32) {
    // S = q @ k^T  (16 x 32), two 16x16 C-frags
    v8f s0 = {}, s1 = {};
    const bf16* kb0 = kbb + (size_t)kt * HDP;
    const bf16* kb1 = kbb + (size_t)(kt + 16) * HDP;
#pragma unroll
    for (int c = 0; c < 3; ++c) {
      v16bf bk0 = load_b_frag(kb0 + c * 32, HDP);
      v16bf bk1 = load_b_frag(kb1 + c * 32, HDP);
      s0 = WMMA_BF16(qf[c], bk0, s0);
      s1 = WMMA_BF16(qf[c], bk1, s1);
    }
    // online softmax (rows grouped per 16-lane half; xor masks 1..8 stay in-half)
    float alpha[8];
#pragma unroll
    for (int r = 0; r < 8; ++r) {
      float mx = fmaxf(s0[r], s1[r]);
      mx = fmaxf(mx, __shfl_xor(mx, 1, 32));
      mx = fmaxf(mx, __shfl_xor(mx, 2, 32));
      mx = fmaxf(mx, __shfl_xor(mx, 4, 32));
      mx = fmaxf(mx, __shfl_xor(mx, 8, 32));
      const float nm = fmaxf(rmax[r], mx);
      alpha[r] = __expf(rmax[r] - nm);
      rmax[r] = nm;
      const float p0 = __expf(s0[r] - nm);
      const float p1 = __expf(s1[r] - nm);
      s0[r] = p0; s1[r] = p1;
      float rs = p0 + p1;
      rs += __shfl_xor(rs, 1, 32);
      rs += __shfl_xor(rs, 2, 32);
      rs += __shfl_xor(rs, 4, 32);
      rs += __shfl_xor(rs, 8, 32);
      rsum[r] = rsum[r] * alpha[r] + rs;
    }
#pragma unroll
    for (int t = 0; t < 6; ++t)
#pragma unroll
      for (int r = 0; r < 8; ++r) o[t][r] *= alpha[r];

    // P: C-layout (f32) -> A-layout (bf16) via per-wave LDS tile
    __syncthreads();
    {
      const int mrow = (lane >> 4) * 8;
      const int ncol = lane & 15;
#pragma unroll
      for (int r = 0; r < 8; ++r) {
        pstage[w][(mrow + r) * 32 + ncol]      = (bf16)s0[r];
        pstage[w][(mrow + r) * 32 + 16 + ncol] = (bf16)s1[r];
      }
    }
    __syncthreads();
    v16bf pf = load_a_frag(&pstage[w][0], 32);

    // O += P @ V  (V^T layout -> contiguous B-frags)
#pragma unroll
    for (int t = 0; t < 6; ++t) {
      v16bf vf = load_b_frag(vbb + (size_t)(t * 16) * SEQ + kt, SEQ);
      o[t] = WMMA_BF16(pf, vf, o[t]);
    }
  }

  // normalize and write [B, N, H*88] in bf16 (proj GEMM input)
  const int rbase = (lane >> 4) * 8;
  const int ncol = lane & 15;
#pragma unroll
  for (int t = 0; t < 6; ++t) {
    const int d = t * 16 + ncol;
    if (d < HD) {
#pragma unroll
      for (int r = 0; r < 8; ++r) {
        const int n = qt * 16 + rbase + r;
        outb[((size_t)b * SEQ + n) * DIMC + h * HD + d] = (bf16)(o[t][r] / rsum[r]);
      }
    }
  }
}

// ---------- host launch ----------
extern "C" void kernel_launch(void* const* d_in, const int* in_sizes, int n_in,
                              void* d_out, int out_size, void* d_ws, size_t ws_size,
                              hipStream_t stream) {
  (void)in_sizes; (void)n_in; (void)out_size; (void)ws_size;
  const float* x      = (const float*)d_in[0];
  const float* w_qkv  = (const float*)d_in[1];
  const float* w_proj = (const float*)d_in[2];
  const float* b_proj = (const float*)d_in[3];
  const float* qw     = (const float*)d_in[4];
  const float* kw     = (const float*)d_in[5];
  float* out = (float*)d_out;
  char* ws = (char*)d_ws;

  const size_t n_x     = (size_t)BATCH * SEQ * DIMC;   // 11,534,336
  const size_t n_wqkv  = (size_t)3 * DIMC * DIMC;      //  5,947,392
  const size_t n_wproj = (size_t)DIMC * DIMC;          //  1,982,464
  const size_t n_head  = (size_t)BATCH * HEADS * SEQ * HDP;

  const size_t OFF_XBF   = 0;
  const size_t OFF_WQKV  = OFF_XBF + n_x * 2;
  const size_t OFF_WPROJ = OFF_WQKV + n_wqkv * 2;
  const size_t OFF_QKV   = OFF_WPROJ + n_wproj * 2;
  const size_t OFF_QS    = OFF_QKV + (size_t)BATCH * SEQ * 3 * DIMC * 4;
  const size_t OFF_KS    = OFF_QS + n_head * 2;
  const size_t OFF_VT    = OFF_KS + n_head * 2;

  bf16* x_bf     = (bf16*)(ws + OFF_XBF);
  bf16* wqkv_bf  = (bf16*)(ws + OFF_WQKV);
  bf16* wproj_bf = (bf16*)(ws + OFF_WPROJ);
  float* qkv_f   = (float*)(ws + OFF_QKV);
  bf16* qs       = (bf16*)(ws + OFF_QS);
  bf16* ksb      = (bf16*)(ws + OFF_KS);
  bf16* vtb      = (bf16*)(ws + OFF_VT);
  bf16* attn_bf  = x_bf;   // x_bf dead after QKV GEMM; same size, reuse

  cvt_f32_bf16<<<(unsigned)((n_x + 255) / 256), 256, 0, stream>>>(x, x_bf, (int)n_x);
  cvt_f32_bf16<<<(unsigned)((n_wqkv + 255) / 256), 256, 0, stream>>>(w_qkv, wqkv_bf, (int)n_wqkv);
  cvt_f32_bf16<<<(unsigned)((n_wproj + 255) / 256), 256, 0, stream>>>(w_proj, wproj_bf, (int)n_wproj);

  // qkv = x @ w_qkv^T : M=8192, N=4224, K=1408  (block tile 128x128)
  dim3 g1(4224 / 128, 8192 / 128);
  gemm_bf16_wmma<<<g1, 256, 0, stream>>>(x_bf, wqkv_bf, qkv_f, nullptr,
                                         DIMC, DIMC, DIMC, 3 * DIMC);

  rope_rms_kernel<<<(BATCH * HEADS * SEQ) / 8, 256, 0, stream>>>(qkv_f, qw, kw, qs, ksb, vtb);

  attn_wmma<<<(BATCH * HEADS * (SEQ / 16)) / 8, 256, 0, stream>>>(qs, ksb, vtb, attn_bf);

  // out = attn @ w_proj^T + b : M=8192, N=1408, K=1408
  dim3 g2(1408 / 128, 8192 / 128);
  gemm_bf16_wmma<<<g2, 256, 0, stream>>>(attn_bf, wproj_bf, out, b_proj,
                                         DIMC, DIMC, DIMC, DIMC);
}
